// StochasticLSTM_3246995276171
// MI455X (gfx1250) — matrix operand order repaired and checked by reading, not compiled, not run on hardware
//
#include <hip/hip_runtime.h>
#include <hip/hip_bf16.h>
#include <stdint.h>

// ---------------- problem constants ----------------
#define BATCH   2048
#define SEQLEN  2048
#define DSIZE   8
#define HSIZE   256
#define GDIM    1024          // 4*HSIZE
#define KG      288           // padded gate K: [o(8) | pad(24) | h(256)]
#define KT_G    9             // 288/32 k-tiles for gates GEMM
#define MTILE   32            // batch rows per workgroup
#define NWG     (BATCH / MTILE)   // 64 workgroups

// packed-weight workspace layout (bytes)
#define WG_ELEMS   (KT_G * GDIM * 32)        // 294912 bf16
#define W1P_OFF    (WG_ELEMS * 2)            // 589824
#define W1P_ELEMS  (8 * 112 * 32)            // 28672 bf16 (K=256 -> 8 kt, N pad 112)
#define TAIL_OFF   (W1P_OFF + W1P_ELEMS * 2) // 647168
#define W2P_ELEMS  (4 * 16 * 32)             // 2048 bf16 (K pad 128 -> 4 kt, N=16)
#define TAIL_BYTES (W2P_ELEMS * 2 + 112 * 4 + 16 * 4)  // 4608 = 1152 dwords

typedef __bf16 bf16;
typedef __attribute__((ext_vector_type(16))) __bf16 v16bf;
typedef __attribute__((ext_vector_type(8)))  float  v8f;
typedef __attribute__((ext_vector_type(4)))  unsigned int u32x4;
typedef __attribute__((ext_vector_type(8)))  int         i32x8;
typedef __attribute__((ext_vector_type(4)))  int         i32x4;

struct Frag32B { uint4 lo, hi; };   // 32 bytes == one v16bf

__device__ __forceinline__ v16bf make_frag(uint4 lo, uint4 hi) {
    Frag32B f{lo, hi};
    return __builtin_bit_cast(v16bf, f);
}

__device__ __forceinline__ v8f v8f_zero() {
    v8f z;
#pragma unroll
    for (int i = 0; i < 8; ++i) z[i] = 0.0f;
    return z;
}

// A-fragment (contiguous k order, same as packed B): lane<16 -> K {kt*32+0..15}
//                                     lane>=16 -> {kt*32+16..31}
template <int ROWSTRIDE>
__device__ __forceinline__ v16bf load_a_frag(const bf16* base, int m, int kt, int hk) {
    const bf16* r = base + m * ROWSTRIDE + kt * 32 + hk * 16;
    const uint4* p0 = reinterpret_cast<const uint4*>(r);
    const uint4* p1 = reinterpret_cast<const uint4*>(r + 8);
    return make_frag(p0[0], p1[0]);
}

// B-fragment from packed [kt][n][khalf][16] storage: 32 contiguous bytes per lane
__device__ __forceinline__ v16bf load_b_frag(const bf16* packed, int elem_off) {
    const uint4* p = reinterpret_cast<const uint4*>(packed + elem_off);
    return make_frag(p[0], p[1]);
}

__device__ __forceinline__ v8f wmma_bf16(v16bf a, v16bf b, v8f c) {
    return __builtin_amdgcn_wmma_f32_16x16x32_bf16(false, a, false, b, (short)0, c,
                                                   false, false);
}

__device__ __forceinline__ float sigmoid_f(float x) { return 1.0f / (1.0f + __expf(-x)); }
__device__ __forceinline__ float softplus_f(float x) {
    return fmaxf(x, 0.0f) + log1pf(__expf(-fabsf(x)));
}

// ---------------- weight packing kernels ----------------
// gates: combined K=288 weight [Wif(8) ; zero(24) ; Whf(256)] -> bf16 packed
__global__ void pack_gate_w(const float* __restrict__ Wif, const float* __restrict__ Whf,
                            bf16* __restrict__ wg) {
    int idx = blockIdx.x * 256 + threadIdx.x;
    if (idx >= WG_ELEMS) return;
    int e2 = idx & 31;
    int rest = idx >> 5;
    int n = rest & (GDIM - 1);
    int kt = rest >> 10;
    int K = kt * 32 + e2;
    float v = 0.0f;
    if (K < DSIZE)                 v = Wif[K * GDIM + n];
    else if (K >= 32 && K < KG)    v = Whf[(K - 32) * GDIM + n];
    wg[idx] = (bf16)v;
}

// W1^T packed: B1[k][n] = (n<100) ? W1[n][k] : 0 ; K=256 (8 kt), N=112
__global__ void pack_w1(const float* __restrict__ W1, bf16* __restrict__ w1p) {
    int idx = blockIdx.x * 256 + threadIdx.x;
    if (idx >= W1P_ELEMS) return;
    int e2 = idx & 31;
    int rest = idx >> 5;
    int n = rest % 112;
    int kt = rest / 112;
    int K = kt * 32 + e2;
    float v = (n < 100) ? W1[n * HSIZE + K] : 0.0f;
    w1p[idx] = (bf16)v;
}

// tail block: W2^T packed (K pad 128, N=16) + b1 padded f32[112] + b2 f32[16]
__global__ void pack_tail(const float* __restrict__ W2, const float* __restrict__ b1,
                          const float* __restrict__ b2, unsigned char* __restrict__ tail) {
    int idx = blockIdx.x * 256 + threadIdx.x;
    bf16* w2p = reinterpret_cast<bf16*>(tail);
    float* b1p = reinterpret_cast<float*>(tail + W2P_ELEMS * 2);
    float* b2p = reinterpret_cast<float*>(tail + W2P_ELEMS * 2 + 112 * 4);
    if (idx < W2P_ELEMS) {
        int e2 = idx & 31;
        int rest = idx >> 5;
        int n = rest & 15;
        int kt = rest >> 4;
        int K = kt * 32 + e2;
        float v = (K < 100) ? W2[n * 100 + K] : 0.0f;
        w2p[idx] = (bf16)v;
    } else if (idx < W2P_ELEMS + 112) {
        int i = idx - W2P_ELEMS;
        b1p[i] = (i < 100) ? b1[i] : 0.0f;
    } else if (idx < W2P_ELEMS + 112 + 16) {
        b2p[idx - W2P_ELEMS - 112] = b2[idx - W2P_ELEMS - 112];
    }
}

// ---------------- persistent recurrent kernel ----------------
__global__ __launch_bounds__(256, 1)
void lstm_main(const float* __restrict__ h0, const float* __restrict__ c0,
               const float* __restrict__ eps0, const float* __restrict__ eps,
               const unsigned char* __restrict__ wsb, float* __restrict__ out) {
    const bf16* wg  = reinterpret_cast<const bf16*>(wsb);
    const bf16* w1p = reinterpret_cast<const bf16*>(wsb + W1P_OFF);

    // LDS: A-matrix rows [o(8)|pad(24)|h(256)] bf16; s1 (32x128) bf16; tail; z f32
    __shared__ bf16  sh_hA[MTILE * KG];            // 18432 B
    __shared__ bf16  sh_s1[MTILE * 128];           //  8192 B
    __shared__ alignas(16) unsigned char sh_tail[TAIL_BYTES]; // 4608 B (TDM dest)
    __shared__ float sh_z[MTILE * 16];             //  2048 B

    const bf16*  sw2 = reinterpret_cast<const bf16*>(sh_tail);
    const float* sb1 = reinterpret_cast<const float*>(sh_tail + W2P_ELEMS * 2);
    const float* sb2 = reinterpret_cast<const float*>(sh_tail + W2P_ELEMS * 2 + 112 * 4);

    const int tid  = threadIdx.x;
    const int lane = tid & 31;
    const int w    = tid >> 5;      // wave 0..7
    const int l15  = lane & 15;
    const int hk   = lane >> 4;     // lane half
    const int wgbase = blockIdx.x * MTILE;

    // --- prologue: TDM bulk copy of W2 + biases into LDS (one tensor DMA) ---
#if __has_builtin(__builtin_amdgcn_tensor_load_to_lds)
    if (w == 0) {
        unsigned long long ga = (unsigned long long)(uintptr_t)(wsb + TAIL_OFF);
        unsigned int lds = (unsigned int)(uintptr_t)(void*)sh_tail;
        u32x4 g0 = { 1u /*count=1*/, lds,
                     (unsigned int)(ga & 0xffffffffull),
                     (unsigned int)(((ga >> 32) & 0x01ffffffull) | 0x80000000u) /*type=2*/ };
        // data_size=4B (code 2); tensor_dim0 = tile_dim0 = 1152 dwords; 1-row tile
        i32x8 g1 = { 0x00020000, (int)(1152u << 16), 0x00010000, (int)(1152u << 16),
                     1, 1152, 0, 0 };
        i32x4 gz4 = { 0, 0, 0, 0 };
        i32x8 gz8 = { 0, 0, 0, 0, 0, 0, 0, 0 };
        __builtin_amdgcn_tensor_load_to_lds(g0, g1, gz4, gz4, gz8, 0);
#if __has_builtin(__builtin_amdgcn_s_wait_tensorcnt)
        __builtin_amdgcn_s_wait_tensorcnt(0);
#endif
    }
#else
    for (int i = tid; i < TAIL_BYTES / 4; i += 256)
        reinterpret_cast<unsigned int*>(sh_tail)[i] =
            reinterpret_cast<const unsigned int*>(wsb + TAIL_OFF)[i];
#endif

    // h0 -> hA[:,32:288) bf16 ; zero pad columns 8..31 ; zero s1
    for (int i = tid; i < MTILE * HSIZE; i += 256) {
        int m = i >> 8, k = i & 255;
        sh_hA[m * KG + 32 + k] = (bf16)h0[(size_t)(wgbase + m) * HSIZE + k];
    }
    for (int i = tid; i < MTILE * 24; i += 256) {
        int m = i / 24, k = i % 24;
        sh_hA[m * KG + 8 + k] = (bf16)0.0f;
    }
    for (int i = tid; i < MTILE * 128; i += 256) sh_s1[i] = (bf16)0.0f;
    __syncthreads();

    // c state in registers, same (m,col) mapping as WMMA C/D fragments:
    // wave w owns h/c columns [32w, 32w+32); j selects 16-col half; m = 16*mt+8*hk+v
    v8f cfr[2][2];
#pragma unroll
    for (int j = 0; j < 2; ++j)
#pragma unroll
        for (int mt = 0; mt < 2; ++mt)
#pragma unroll
            for (int v = 0; v < 8; ++v) {
                int m = 16 * mt + 8 * hk + v;
                int hcol = 32 * w + 16 * j + l15;
                cfr[j][mt][v] = c0[(size_t)(wgbase + m) * HSIZE + hcol];
            }

    // ---- sampling head: h(LDS) @ W1^T -> softplus -> @ W2^T -> softplus -> o ----
    auto do_sample = [&](const float* eps_base, float* outp, int t) {
        // GEMM1: M=32, N=112 (7 tiles), K=256 ; 14 output tiles over 8 waves
#pragma unroll
        for (int rep = 0; rep < 2; ++rep) {
            int T = w + 8 * rep;
            if (T < 14) {
                int mt = T / 7, nt = T % 7;
                int n1 = nt * 16 + l15;           // < 112
                v8f acc = v8f_zero();
                for (int kt = 0; kt < 8; ++kt) {
                    v16bf a = load_a_frag<KG>(sh_hA, l15 + 16 * mt, kt + 1, hk);
                    v16bf b = load_b_frag(w1p, (kt * 112 + n1) * 32 + hk * 16);
                    acc = wmma_bf16(a, b, acc);
                }
#pragma unroll
                for (int v = 0; v < 8; ++v) {
                    int m = 16 * mt + 8 * hk + v;
                    float z1 = softplus_f(acc[v] + sb1[n1]);
                    sh_s1[m * 128 + n1] = (bf16)z1;
                }
            }
        }
        __syncthreads();
        // GEMM2: M=32, N=16, K=128 ; waves 0,1 each take one M-tile
        if (w < 2) {
            int mt = w;
            v8f acc = v8f_zero();
#pragma unroll
            for (int kt = 0; kt < 4; ++kt) {
                v16bf a = load_a_frag<128>(sh_s1, l15 + 16 * mt, kt, hk);
                v16bf b = load_b_frag(sw2, (kt * 16 + l15) * 32 + hk * 16);
                acc = wmma_bf16(a, b, acc);
            }
#pragma unroll
            for (int v = 0; v < 8; ++v) {
                int m = 16 * mt + 8 * hk + v;
                sh_z[m * 16 + l15] = softplus_f(acc[v] + sb2[l15]);
            }
        }
        __syncthreads();
        // o = softplus(mean + sqrt(var)*eps) ; 256 threads <-> 32x8 elements
        {
            int m = tid >> 3, d = tid & 7;
            int b = wgbase + m;
            float mean = sh_z[m * 16 + d];
            float var  = sh_z[m * 16 + 8 + d];
            float e    = eps_base[(size_t)b * DSIZE + d];
            float o    = softplus_f(mean + sqrtf(var) * e);
            sh_hA[m * KG + d] = (bf16)o;      // A-matrix o-columns for next step
            if (outp) outp[((size_t)b * SEQLEN + t) * DSIZE + d] = o;
        }
        __syncthreads();
    };

    // o0 = sample(h0, eps0): seeds gate input, not written to output
    do_sample(eps0, nullptr, 0);

    // ---------------- time loop ----------------
    for (int t = 0; t < SEQLEN; ++t) {
        // gates GEMM: [o|h](32xK288) @ Wg(288x1024); wave w owns N-tiles
        // nt = q*16 + 2w + j  (q = gate, so i/f/g/og columns stay wave-local)
        v8f acc[4][2][2];
#pragma unroll
        for (int q = 0; q < 4; ++q)
#pragma unroll
            for (int j = 0; j < 2; ++j)
#pragma unroll
                for (int mt = 0; mt < 2; ++mt) acc[q][j][mt] = v8f_zero();

        for (int kt = 0; kt < KT_G; ++kt) {
            if (kt + 1 < KT_G) {   // stream next weight k-tile toward L2/L0
                int npf = (2 * w) * 16 + l15;
                __builtin_prefetch(wg + ((kt + 1) * GDIM + npf) * 32 + hk * 16, 0, 1);
            }
            v16bf a0 = load_a_frag<KG>(sh_hA, l15, kt, hk);
            v16bf a1 = load_a_frag<KG>(sh_hA, l15 + 16, kt, hk);
#pragma unroll
            for (int q = 0; q < 4; ++q)
#pragma unroll
                for (int j = 0; j < 2; ++j) {
                    int nt = q * 16 + 2 * w + j;
                    int n  = nt * 16 + l15;
                    v16bf b = load_b_frag(wg, (kt * GDIM + n) * 32 + hk * 16);
                    acc[q][j][0] = wmma_bf16(a0, b, acc[q][j][0]);
                    acc[q][j][1] = wmma_bf16(a1, b, acc[q][j][1]);
                }
        }
        __syncthreads();   // all A-reads of hA done before h_new overwrite

        // LSTM cell (wave-local: all four gates for owned columns)
#pragma unroll
        for (int j = 0; j < 2; ++j)
#pragma unroll
            for (int mt = 0; mt < 2; ++mt)
#pragma unroll
                for (int v = 0; v < 8; ++v) {
                    float gi = acc[0][j][mt][v];
                    float gf = acc[1][j][mt][v];
                    float gg = acc[2][j][mt][v];
                    float go = acc[3][j][mt][v];
                    float c  = cfr[j][mt][v];
                    float cn = sigmoid_f(gf) * c + sigmoid_f(gi) * tanhf(gg);
                    float hn = sigmoid_f(go) * tanhf(cn);
                    cfr[j][mt][v] = cn;
                    int m = 16 * mt + 8 * hk + v;
                    int hcol = 32 * w + 16 * j + l15;
                    sh_hA[m * KG + 32 + hcol] = (bf16)hn;
                }
        __syncthreads();   // h_new visible to all waves

        do_sample(eps + (size_t)t * BATCH * DSIZE, out, t);
    }
}

// ---------------- host launcher ----------------
extern "C" void kernel_launch(void* const* d_in, const int* in_sizes, int n_in,
                              void* d_out, int out_size, void* d_ws, size_t ws_size,
                              hipStream_t stream) {
    const float* h0   = (const float*)d_in[0];
    const float* c0   = (const float*)d_in[1];
    const float* eps0 = (const float*)d_in[2];
    const float* eps  = (const float*)d_in[3];
    const float* Wif  = (const float*)d_in[4];
    const float* Whf  = (const float*)d_in[5];
    const float* W1   = (const float*)d_in[6];
    const float* b1   = (const float*)d_in[7];
    const float* W2   = (const float*)d_in[8];
    const float* b2   = (const float*)d_in[9];
    unsigned char* ws = (unsigned char*)d_ws;
    float* out = (float*)d_out;

    pack_gate_w<<<(WG_ELEMS + 255) / 256, 256, 0, stream>>>(Wif, Whf, (bf16*)ws);
    pack_w1<<<(W1P_ELEMS + 255) / 256, 256, 0, stream>>>(W1, (bf16*)(ws + W1P_OFF));
    pack_tail<<<9, 256, 0, stream>>>(W2, b1, b2, ws + TAIL_OFF);
    lstm_main<<<NWG, 256, 0, stream>>>(h0, c0, eps0, eps, ws, out);
}
